// ASTEmbeddings_14851996909620
// MI455X (gfx1250) — matrix operand order repaired and checked by reading, not compiled, not run on hardware
//
#include <hip/hip_runtime.h>
#include <hip/hip_bf16.h>

// Problem constants (from reference)
#define BB 64
#define NN 1024
#define DD 256
#define HH 256
#define TH3 768   // 3*H

typedef __bf16 v16bf __attribute__((ext_vector_type(16)));
typedef float  v8f   __attribute__((ext_vector_type(8)));

union V16 {
    v16bf b;
    unsigned short u[16];
    uint4 q[2];
};

__device__ __forceinline__ unsigned short bf16_rn(float f) {
    unsigned int x = __float_as_uint(f);
    unsigned int r = (x + 0x7FFFu + ((x >> 16) & 1u)) >> 16;
    return (unsigned short)r;
}
__device__ __forceinline__ float bf16_to_f32(unsigned short u) {
    return __uint_as_float(((unsigned int)u) << 16);
}
__device__ __forceinline__ float sigmoidf_fast(float x) {
    return 1.0f / (1.0f + __expf(-x));
}
__device__ __forceinline__ float tanhf_fast(float x) {
    float t = __expf(2.0f * x);
    return (t - 1.0f) / (t + 1.0f);
}
__device__ __forceinline__ uint4 pack8(const unsigned short* u) {
    uint4 q;
    q.x = (unsigned)u[0] | ((unsigned)u[1] << 16);
    q.y = (unsigned)u[2] | ((unsigned)u[3] << 16);
    q.z = (unsigned)u[4] | ((unsigned)u[5] << 16);
    q.w = (unsigned)u[6] | ((unsigned)u[7] << 16);
    return q;
}

// ---------------------------------------------------------------------------
// Kernel 1: node_type_embed = type_emb[t]*4 + pos_table[cp]/4  -> xws (fp32)
//           out[n,b,0:256]  = node_type_embed + EmbeddingBag_mean * 4
// grid: B*N blocks, 256 threads (thread = d)
// ---------------------------------------------------------------------------
__global__ __launch_bounds__(256) void k_embed_bag(
    const int* __restrict__ node_types, const int* __restrict__ node_vals,
    const int* __restrict__ node_val_offsets, const int* __restrict__ child_positions,
    const float* __restrict__ type_emb, const float* __restrict__ pos_table,
    const float* __restrict__ token_emb, float* __restrict__ xws,
    float* __restrict__ out, int Lvals)
{
    int bag = blockIdx.x;             // b*N + n
    int d = threadIdx.x;
    int b = bag >> 10, n = bag & (NN - 1);
    int t  = node_types[bag];
    int cp = child_positions[bag];
    float nte = type_emb[(size_t)t * DD + d] * 4.0f
              + pos_table[(size_t)cp * DD + d] * 0.25f;
    xws[(size_t)bag * DD + d] = nte;

    int off = node_val_offsets[bag];
    int end = (bag + 1 < BB * NN) ? node_val_offsets[bag + 1] : Lvals;
    float s = 0.0f;
    for (int v = off; v < end; ++v)
        s += token_emb[(size_t)node_vals[v] * DD + d];
    int cnt = end - off;
    float mean = s / (float)(cnt > 1 ? cnt : 1);
    out[((size_t)n * BB + b) * (DD + HH) + d] = nte + mean * 4.0f;
}

// ---------------------------------------------------------------------------
// Kernel 2: pack w_ih / w_hh (fp32 [768,256] row-major) into bf16 WMMA B-layout
// Packed flat index: (((nt*8 + kt)*32 + lane)*16 + e)
//   lane<16 : N = nt*16 + lane,    K = kt*32 + e (e<8), kt*32 + 8 + e (e>=8)
//   lane>=16: N = nt*16 + lane-16, K = kt*32 + 8 + e (e<8), kt*32 + 16 + e (e>=8)
// ---------------------------------------------------------------------------
__global__ __launch_bounds__(256) void k_pack_w(
    const float* __restrict__ w_ih, const float* __restrict__ w_hh,
    unsigned short* __restrict__ wih_bf, unsigned short* __restrict__ whh_bf)
{
    int tid = blockIdx.x * 256 + threadIdx.x;   // ((nt*8)+kt)*32 + lane
    if (tid >= 48 * 8 * 32) return;
    int lane = tid & 31;
    const float* W = blockIdx.y ? w_hh : w_ih;
    unsigned short* Dst = blockIdx.y ? whh_bf : wih_bf;
    int nt = tid >> 8, kt = (tid >> 5) & 7;
    int ncol = nt * 16 + (lane & 15);                 // row of W (= col of W^T)
    int kbase = kt * 32 + (lane < 16 ? 0 : 8);
    unsigned short* p = Dst + (size_t)tid * 16;
    #pragma unroll
    for (int e = 0; e < 16; ++e)
        p[e] = bf16_rn(W[(size_t)ncol * HH + kbase + e + (e >= 8 ? 8 : 0)]);
}

// ---------------------------------------------------------------------------
// Kernel 3: gi[65536, 768] = x[65536,256] @ w_ih^T + b_ih   (bf16 WMMA, f32 acc)
// 256 thr = 8 waves/block; wave computes 16 rows x 192 cols (12 n-tiles), K=256.
// Software-pipelined: B fragments for kt+1 are loaded while kt's WMMAs issue.
// ---------------------------------------------------------------------------
__device__ __forceinline__ void loadB(V16& dst, const unsigned short* __restrict__ wbf,
                                      int nt, int kt, int lane)
{
    const unsigned short* p = wbf + ((((size_t)nt * 8 + kt) * 32 + lane) * 16);
    dst.q[0] = *(const uint4*)p;
    dst.q[1] = *(const uint4*)(p + 8);
}

__global__ __launch_bounds__(256) void k_gi_gemm(
    const float* __restrict__ xws, const unsigned short* __restrict__ wih_bf,
    const float* __restrict__ b_ih, float* __restrict__ gi)
{
    int wave = threadIdx.x >> 5, lane = threadIdx.x & 31;
    int gw = blockIdx.x * 8 + wave;
    int mtile = gw >> 2, cg = gw & 3;
    int m0 = mtile * 16;
    int ntbase = cg * 12;

    v8f acc[12];
    #pragma unroll
    for (int t = 0; t < 12; ++t)
        #pragma unroll
        for (int e = 0; e < 8; ++e) acc[t][e] = 0.0f;

    const float* arow = xws + (size_t)(m0 + (lane & 15)) * HH;
    int koffA = (lane < 16) ? 0 : 8;

    // Double-buffered B fragments (w_ih is L2-resident: 384 KB shared by all blocks)
    V16 bfr[2][12];
    #pragma unroll
    for (int t = 0; t < 12; ++t) loadB(bfr[0][t], wih_bf, ntbase + t, 0, lane);

    #pragma unroll
    for (int kt = 0; kt < 8; ++kt) {
        int cur = kt & 1;
        // Build bf16 A fragment (16x32) from fp32 x: K runs {base..+7, base+16..+23}
        const float* pA = arow + kt * 32 + koffA;
        float fa[16];
        *(float4*)&fa[0]  = *(const float4*)(pA);
        *(float4*)&fa[4]  = *(const float4*)(pA + 4);
        *(float4*)&fa[8]  = *(const float4*)(pA + 16);
        *(float4*)&fa[12] = *(const float4*)(pA + 20);
        V16 a;
        #pragma unroll
        for (int e = 0; e < 16; ++e) a.u[e] = bf16_rn(fa[e]);

        if (kt < 7) {
            #pragma unroll
            for (int t = 0; t < 12; ++t)
                loadB(bfr[cur ^ 1][t], wih_bf, ntbase + t, kt + 1, lane);
        }

        #pragma unroll
        for (int t = 0; t < 12; ++t)
            acc[t] = __builtin_amdgcn_wmma_f32_16x16x32_bf16(
                false, a.b, false, bfr[cur][t].b, (short)0, acc[t], false, false);
    }

    #pragma unroll
    for (int t = 0; t < 12; ++t) {
        int col = (ntbase + t) * 16 + (lane & 15);
        float bias = b_ih[col];
        #pragma unroll
        for (int r = 0; r < 8; ++r) {
            int row = m0 + r + (lane >= 16 ? 8 : 0);
            gi[(size_t)row * TH3 + col] = acc[t][r] + bias;
        }
    }
}

// ---------------------------------------------------------------------------
// Kernel 4: sequential tree-GRU scan. ONE persistent workgroup, 512 thr (16 waves).
// w_hh lives in VGPRs (bf16 B-layout) for all 1024 steps. Each step:
//   A) gather h_prev[b] = out[lpi[b,i], b, 256:] -> LDS (bf16, WMMA A-layout)
//      + prefetch this step's gi rows (covers phase C, hidden under the WMMAs)
//   B) gh = h_prev @ w_hh^T via 96 WMMA/wave (wave w owns cols 16w.. of all 3 gates)
//   C) register-local GRU epilogue, store h -> out[i, b, 256:]
// ---------------------------------------------------------------------------
__global__ __launch_bounds__(512, 1) void k_gru_seq(
    const float* __restrict__ gi, const unsigned short* __restrict__ whh_bf,
    const float* __restrict__ b_hh, const int* __restrict__ last_parent_index,
    float* out)
{
    __shared__ unsigned short Ash[4][8][32][16];   // 32 KB: A-frags, [mtile][ktile][lane][e]

    int tid = threadIdx.x;
    int wave = tid >> 5, lane = tid & 31;

    // Resident weights: wave w handles n-tiles {w, w+16, w+32} = gates r,z,n cols 16w..16w+15
    V16 wB[3][8];
    #pragma unroll
    for (int g = 0; g < 3; ++g)
        #pragma unroll
        for (int kt = 0; kt < 8; ++kt) {
            const unsigned short* p =
                whh_bf + ((((size_t)(wave + 16 * g) * 8 + kt) * 32 + lane) * 16);
            wB[g][kt].q[0] = *(const uint4*)p;
            wB[g][kt].q[1] = *(const uint4*)(p + 8);
        }

    int jcol = wave * 16 + (lane & 15);
    float bhh_r = b_hh[jcol];
    float bhh_z = b_hh[jcol + HH];
    float bhh_n = b_hh[jcol + 2 * HH];

    // Phase-A thread mapping: 512 thr -> (b = tid/8, ktile = tid%8), 32 k each
    int ga_b = tid >> 3, ga_kt = tid & 7;
    int ga_mt = ga_b >> 4, ga_mr = ga_b & 15;

    // h_prev LDS lookup constants for the epilogue (b decided later; j fixed per lane)
    int ep_koff = jcol & 31, ep_ktl = jcol >> 5;
    int ep_lhi = ((ep_koff >> 3) & 1) ? 16 : 0;
    int ep_e   = (ep_koff & 7) + ((ep_koff >= 16) ? 8 : 0);

    for (int i = 0; i < NN; ++i) {
        // ---- Phase A: gather + bf16-convert h_prev into WMMA A-layout LDS ----
        {
            // Prefetch this step's gi row-set (b covers all 64 rows across threads;
            // 8 threads/row x 96 floats each = full 3KB row). Lands during phase B.
            const float* gip = gi + ((size_t)ga_b * NN + i) * TH3 + ga_kt * 96;
            __builtin_prefetch(gip,      0, 1);
            __builtin_prefetch(gip + 32, 0, 1);
            __builtin_prefetch(gip + 64, 0, 1);

            float f[32];
            if (i == 0) {
                #pragma unroll
                for (int e = 0; e < 32; ++e) f[e] = 0.0f;
            } else {
                int lpi = last_parent_index[ga_b * NN + i];
                const float* src = out + ((size_t)lpi * BB + ga_b) * (DD + HH) + DD + ga_kt * 32;
                #pragma unroll
                for (int q = 0; q < 8; ++q)
                    *(float4*)&f[q * 4] = *(const float4*)(src + q * 4);
            }
            unsigned short u[32];
            #pragma unroll
            for (int e = 0; e < 32; ++e) u[e] = bf16_rn(f[e]);
            unsigned short* dLo = &Ash[ga_mt][ga_kt][ga_mr][0];
            unsigned short* dHi = &Ash[ga_mt][ga_kt][ga_mr + 16][0];
            *(uint4*)(dLo)     = pack8(u);        // K 0..7   -> e 0..7,  lanes 0-15
            *(uint4*)(dLo + 8) = pack8(u + 16);   // K 16..23 -> e 8..15, lanes 0-15
            *(uint4*)(dHi)     = pack8(u + 8);    // K 8..15  -> e 0..7,  lanes 16-31
            *(uint4*)(dHi + 8) = pack8(u + 24);   // K 24..31 -> e 8..15, lanes 16-31
        }
        __syncthreads();

        // ---- Phase B: gh accumulation, 96 WMMA per wave ----
        v8f acc[3][4];
        #pragma unroll
        for (int g = 0; g < 3; ++g)
            #pragma unroll
            for (int m = 0; m < 4; ++m)
                #pragma unroll
                for (int e = 0; e < 8; ++e) acc[g][m][e] = 0.0f;

        #pragma unroll
        for (int kt = 0; kt < 8; ++kt) {
            V16 a[4];
            #pragma unroll
            for (int m = 0; m < 4; ++m) {
                const unsigned short* p = &Ash[m][kt][lane][0];
                a[m].q[0] = *(const uint4*)p;
                a[m].q[1] = *(const uint4*)(p + 8);
            }
            #pragma unroll
            for (int g = 0; g < 3; ++g)
                #pragma unroll
                for (int m = 0; m < 4; ++m)
                    acc[g][m] = __builtin_amdgcn_wmma_f32_16x16x32_bf16(
                        false, a[m].b, false, wB[g][kt].b, (short)0,
                        acc[g][m], false, false);
        }

        // ---- Phase C: GRU nonlinearity (register-local: all 3 gates in this wave) ----
        #pragma unroll
        for (int m = 0; m < 4; ++m) {
            #pragma unroll
            for (int r = 0; r < 8; ++r) {
                int mrow = r + (lane >= 16 ? 8 : 0);
                int b = m * 16 + mrow;
                const float* gr = gi + ((size_t)b * NN + i) * TH3;
                float ir  = gr[jcol];
                float iz  = gr[jcol + HH];
                float inn = gr[jcol + 2 * HH];
                float hp  = bf16_to_f32(Ash[m][ep_ktl][mrow + ep_lhi][ep_e]);
                float rg = sigmoidf_fast(ir + acc[0][m][r] + bhh_r);
                float zg = sigmoidf_fast(iz + acc[1][m][r] + bhh_z);
                float ng = tanhf_fast(inn + rg * (acc[2][m][r] + bhh_n));
                float h  = (1.0f - zg) * ng + zg * hp;
                out[((size_t)i * BB + b) * (DD + HH) + DD + jcol] = h;
            }
        }

        __threadfence();     // make hidden-state stores visible (invalidate WGP$)
        __syncthreads();     // also protects Ash reuse next step
    }
}

// ---------------------------------------------------------------------------
extern "C" void kernel_launch(void* const* d_in, const int* in_sizes, int n_in,
                              void* d_out, int out_size, void* d_ws, size_t ws_size,
                              hipStream_t stream)
{
    const int*   node_types        = (const int*)d_in[0];
    const int*   node_vals         = (const int*)d_in[1];
    const int*   node_val_offsets  = (const int*)d_in[2];
    const int*   last_parent_index = (const int*)d_in[3];
    const int*   child_positions   = (const int*)d_in[4];
    const float* type_emb          = (const float*)d_in[5];
    const float* pos_table         = (const float*)d_in[6];
    const float* token_emb         = (const float*)d_in[7];
    const float* w_ih              = (const float*)d_in[8];
    const float* w_hh              = (const float*)d_in[9];
    const float* b_ih              = (const float*)d_in[10];
    const float* b_hh              = (const float*)d_in[11];
    float* out = (float*)d_out;

    // Workspace layout (requires ~257 MB):
    //  [0)        wih_bf  bf16 packed   384 KB
    //  [384K)     whh_bf  bf16 packed   384 KB
    //  [768K)     xws     fp32 [65536,256]   64 MB
    //  [768K+64M) gi      fp32 [65536,768]  192 MB
    char* ws = (char*)d_ws;
    unsigned short* wih_bf = (unsigned short*)ws;
    unsigned short* whh_bf = (unsigned short*)(ws + 393216);
    float* xws = (float*)(ws + 786432);
    float* gi  = (float*)(ws + 786432 + (size_t)BB * NN * DD * 4);

    k_embed_bag<<<BB * NN, 256, 0, stream>>>(node_types, node_vals, node_val_offsets,
                                             child_positions, type_emb, pos_table,
                                             token_emb, xws, out, in_sizes[1]);
    k_pack_w<<<dim3(48, 2), 256, 0, stream>>>(w_ih, w_hh, wih_bf, whh_bf);
    k_gi_gemm<<<2048, 256, 0, stream>>>(xws, wih_bf, b_ih, gi);
    k_gru_seq<<<1, 512, 0, stream>>>(gi, whh_bf, b_hh, last_parent_index, out);
}
